// CTSMamba_85263690760311
// MI455X (gfx1250) — compile-verified, matrix operand order checked by reading, not run on hardware
//
#include <hip/hip_runtime.h>
#include <hip/hip_bf16.h>

// ---------------- problem constants ----------------
#define Bn   4
#define Cc   384
#define Lx   4096              // D*H*W tokens per batch
#define Mtok (Bn * Lx)         // 16384 total tokens
#define DIN  768
#define DIN2 1536
#define DST  16                // DSTATE
#define DTR  24                // DTRANK
#define DTRP 32                // padded DTRANK (K multiple of 32)
#define DBLW 56                // DTRANK + 2*DSTATE
#define DBLP 64                // padded dbl row stride / x_proj N
#define PR   18                // padded spatial extent (16 + halo)
#define PA   (PR * PR)         // 324
#define PV   (PR * PR * PR)    // 5832 voxels incl halo

typedef _Float16 v16h __attribute__((ext_vector_type(16)));
typedef float    v8f  __attribute__((ext_vector_type(8)));
typedef unsigned u32x4 __attribute__((ext_vector_type(4)));

union Frag { u32x4 q[2]; v16h v; };

__device__ __forceinline__ float sigmoidf_(float x) { return 1.0f / (1.0f + __expf(-x)); }

__device__ __forceinline__ void load_frag(Frag& f, const unsigned* row, int kd) {
    f.q[0] = *(const u32x4*)(row + kd);
    f.q[1] = *(const u32x4*)(row + kd + 8);
}

// ================= elementwise / pack kernels =================

__global__ void k_zero_h(_Float16* __restrict__ p, long n) {
    long i = (long)blockIdx.x * blockDim.x + threadIdx.x;
    if (i < n) p[i] = (_Float16)0.f;
}

// generic f32 -> f16 cast
__global__ void k_cast_h(const float* __restrict__ src, _Float16* __restrict__ dst, long n) {
    long i = (long)blockIdx.x * blockDim.x + threadIdx.x;
    if (i < n) dst[i] = (_Float16)src[i];
}

// zero-padded weight cast: dst[n*Kdst+k] = (n<Nsrc && k<Ksrc) ? src[n*Ksrc+k] : 0
__global__ void k_pad_w(const float* __restrict__ src, _Float16* __restrict__ dst,
                        int Nsrc, int Ksrc, long total, int Kdst) {
    long i = (long)blockIdx.x * blockDim.x + threadIdx.x;
    if (i >= total) return;
    int k = (int)(i % Kdst);
    int n = (int)(i / Kdst);
    dst[i] = (n < Nsrc && k < Ksrc) ? (_Float16)src[(long)n * Ksrc + k] : (_Float16)0.f;
}

// x [B,C,L] f32 -> flat xtf [B,L,C] f16 AND halo-padded xtp [B,18^3,C] f16 interior
__global__ void k_cast_x(const float* __restrict__ x, _Float16* __restrict__ xtf,
                         _Float16* __restrict__ xtp) {
    long i = (long)blockIdx.x * blockDim.x + threadIdx.x;
    long n = (long)Bn * Cc * Lx;
    if (i >= n) return;
    int c = (int)(i % Cc);
    long r = i / Cc;
    int l = (int)(r % Lx);
    int b = (int)(r / Lx);
    _Float16 h = (_Float16)x[((long)b * Cc + c) * Lx + l];
    xtf[i] = h;
    int dz = l >> 8, dy = (l >> 4) & 15, dx = l & 15;
    long pd = (long)(dz + 1) * PA + (dy + 1) * PR + (dx + 1);
    xtp[((long)b * PV + pd) * Cc + c] = h;
}

// gsc 3x3x3 weight (co,ci,27) f32 -> packed (tap,co,ci) f16
__global__ void k_pack_conv(const float* __restrict__ w, _Float16* __restrict__ wp) {
    long i = (long)blockIdx.x * blockDim.x + threadIdx.x;
    long n = 27L * Cc * Cc;
    if (i >= n) return;
    int ci  = (int)(i % Cc);
    long r  = i / Cc;
    int co  = (int)(r % Cc);
    int tap = (int)(r / Cc);
    wp[i] = (_Float16)w[((long)co * Cc + ci) * 27 + tap];
}

// A = -exp(A_log)
__global__ void k_aneg(const float* __restrict__ alog, float* __restrict__ an, int n) {
    int i = blockIdx.x * blockDim.x + threadIdx.x;
    if (i < n) an[i] = -__expf(alog[i]);
}

// f16 elementwise add -> f16
__global__ void k_add_h(const _Float16* __restrict__ a, const _Float16* __restrict__ b,
                        _Float16* __restrict__ o, long n) {
    long i = (long)blockIdx.x * blockDim.x + threadIdx.x;
    if (i < n) o[i] = (_Float16)((float)a[i] + (float)b[i]);
}

// dt = softplus(dt + bias[d]) in-place, dt [Mtok, DIN]
__global__ void k_softplus_bias(float* __restrict__ dt, const float* __restrict__ bias) {
    long i = (long)blockIdx.x * blockDim.x + threadIdx.x;
    long n = (long)Mtok * DIN;
    if (i >= n) return;
    float v = dt[i] + bias[i % DIN];
    dt[i] = (v > 20.0f) ? v : log1pf(__expf(v));
}

// dtin f16 [Mtok, 32] <- dbl f32 [Mtok, 64][:, :24], zero-padded K
__global__ void k_slice_dt(const float* __restrict__ dbl, _Float16* __restrict__ dtin) {
    long i = (long)blockIdx.x * blockDim.x + threadIdx.x;
    long n = (long)Mtok * DTRP;
    if (i >= n) return;
    long m = i / DTRP; int j = (int)(i % DTRP);
    dtin[i] = (j < DTR) ? (_Float16)dbl[m * DBLP + j] : (_Float16)0.f;
}

// ================= WMMA generic GEMM, 2x2 register-blocked =====================
// Requires M%32==0, N%32==0, K%32==0. Each wave computes a 32x32 output block.
// D[m,n] = sum_k A[m,k] * W[n,k]
// store_mode 0: C[m*ldc + n]   store_mode 1: C[((m/Lx)*N + n)*Lx + (m%Lx)]
__global__ void k_gemm_wmma(const _Float16* __restrict__ A, int lda,
                            const _Float16* __restrict__ W, int ldb,
                            float* __restrict__ C, int ldc,
                            int M, int N, int K, int store_mode) {
    int wave = (int)((blockIdx.x * blockDim.x + threadIdx.x) >> 5);
    int lane = threadIdx.x & 31;
    int nt2 = N >> 5;
    int mt2 = M >> 5;
    if (wave >= mt2 * nt2) return;              // wave-uniform: EXEC stays all-1s
    int tm = (wave / nt2) << 5;
    int tn = (wave % nt2) << 5;
    int l15 = lane & 15, kg = lane >> 4;

    const unsigned* arow0 = (const unsigned*)(A + (size_t)(tm + l15) * lda);
    const unsigned* arow1 = arow0 + (size_t)8 * lda;   // +16 rows (dwords)
    const unsigned* brow0 = (const unsigned*)(W + (size_t)(tn + l15) * ldb);
    const unsigned* brow1 = brow0 + (size_t)8 * ldb;

    v8f acc00 = {0.f,0.f,0.f,0.f,0.f,0.f,0.f,0.f};
    v8f acc01 = acc00, acc10 = acc00, acc11 = acc00;
    int ktiles = K >> 5;
    for (int kt = 0; kt < ktiles; ++kt) {
        int kd = (kt << 4) + (kg << 2);         // dword offset of fragment low half
        Frag fa0, fa1, fb0, fb1;
        load_frag(fa0, arow0, kd);
        load_frag(fa1, arow1, kd);
        load_frag(fb0, brow0, kd);
        load_frag(fb1, brow1, kd);
        acc00 = __builtin_amdgcn_wmma_f32_16x16x32_f16(false, fa0.v, false, fb0.v,
                                                       (short)0, acc00, false, false);
        acc01 = __builtin_amdgcn_wmma_f32_16x16x32_f16(false, fa0.v, false, fb1.v,
                                                       (short)0, acc01, false, false);
        acc10 = __builtin_amdgcn_wmma_f32_16x16x32_f16(false, fa1.v, false, fb0.v,
                                                       (short)0, acc10, false, false);
        acc11 = __builtin_amdgcn_wmma_f32_16x16x32_f16(false, fa1.v, false, fb1.v,
                                                       (short)0, acc11, false, false);
    }
#pragma unroll
    for (int v = 0; v < 8; ++v) {
        int m0 = tm + v + (kg << 3);
        int m1 = m0 + 16;
        int n0 = tn + l15;
        int n1 = n0 + 16;
        if (store_mode == 0) {
            C[(size_t)m0 * ldc + n0] = acc00[v];
            C[(size_t)m0 * ldc + n1] = acc01[v];
            C[(size_t)m1 * ldc + n0] = acc10[v];
            C[(size_t)m1 * ldc + n1] = acc11[v];
        } else {                                 // [B,C,L] scatter for out_proj
            int b0 = m0 / Lx, l0 = m0 % Lx;
            int b1 = m1 / Lx, l1 = m1 % Lx;
            C[((size_t)b0 * N + n0) * Lx + l0] = acc00[v];
            C[((size_t)b0 * N + n1) * Lx + l0] = acc01[v];
            C[((size_t)b1 * N + n0) * Lx + l1] = acc10[v];
            C[((size_t)b1 * N + n1) * Lx + l1] = acc11[v];
        }
    }
}

// ================= 3x3x3 conv as implicit GEMM (WMMA, 2x2 blocked) =============
// xp [B, 18^3, C] f16 (halo zeroed), wp [27,Cout,Cin] f16, out [B,L,C] f32 (+bias)
__global__ void k_conv3_wmma(const _Float16* __restrict__ xp,
                             const _Float16* __restrict__ wp,
                             const float* __restrict__ bias,
                             float* __restrict__ out) {
    const int NT2 = Cc / 32;      // 12 n-blocks
    const int MT2 = Lx / 32;      // 128 m-blocks
    int wave = (int)((blockIdx.x * blockDim.x + threadIdx.x) >> 5);
    int lane = threadIdx.x & 31;
    if (wave >= Bn * MT2 * NT2) return;
    int b  = wave / (MT2 * NT2);
    int r  = wave % (MT2 * NT2);
    int tm = (r / NT2) << 5;
    int tn = (r % NT2) << 5;
    int l15 = lane & 15, kg = lane >> 4;

    // two voxel rows per wave (m-tile 0 and 1)
    int mv0 = tm + l15, mv1 = tm + 16 + l15;
    int dz0 = mv0 >> 8, dy0 = (mv0 >> 4) & 15, dx0 = mv0 & 15;
    int dz1 = mv1 >> 8, dy1 = (mv1 >> 4) & 15, dx1 = mv1 & 15;
    long vc0 = (long)b * PV + (long)(dz0 + 1) * PA + (dy0 + 1) * PR + (dx0 + 1);
    long vc1 = (long)b * PV + (long)(dz1 + 1) * PA + (dy1 + 1) * PR + (dx1 + 1);

    v8f acc00 = {0.f,0.f,0.f,0.f,0.f,0.f,0.f,0.f};
    v8f acc01 = acc00, acc10 = acc00, acc11 = acc00;
    for (int tap = 0; tap < 27; ++tap) {
        int kd3 = tap / 9 - 1, kh3 = (tap / 3) % 3 - 1, kw3 = tap % 3 - 1;
        long off = (long)kd3 * PA + kh3 * PR + kw3;
        const unsigned* arow0 = (const unsigned*)(xp + (vc0 + off) * Cc);
        const unsigned* arow1 = (const unsigned*)(xp + (vc1 + off) * Cc);
        const unsigned* brow0 = (const unsigned*)(wp + ((size_t)tap * Cc + tn + l15) * Cc);
        const unsigned* brow1 = brow0 + (size_t)8 * Cc;
        for (int kt = 0; kt < Cc / 32; ++kt) {
            int kd = (kt << 4) + (kg << 2);
            Frag fa0, fa1, fb0, fb1;
            load_frag(fa0, arow0, kd);
            load_frag(fa1, arow1, kd);
            load_frag(fb0, brow0, kd);
            load_frag(fb1, brow1, kd);
            acc00 = __builtin_amdgcn_wmma_f32_16x16x32_f16(false, fa0.v, false, fb0.v,
                                                           (short)0, acc00, false, false);
            acc01 = __builtin_amdgcn_wmma_f32_16x16x32_f16(false, fa0.v, false, fb1.v,
                                                           (short)0, acc01, false, false);
            acc10 = __builtin_amdgcn_wmma_f32_16x16x32_f16(false, fa1.v, false, fb0.v,
                                                           (short)0, acc10, false, false);
            acc11 = __builtin_amdgcn_wmma_f32_16x16x32_f16(false, fa1.v, false, fb1.v,
                                                           (short)0, acc11, false, false);
        }
    }
    float bias0 = bias[tn + l15];
    float bias1 = bias[tn + 16 + l15];
#pragma unroll
    for (int v = 0; v < 8; ++v) {
        int m0 = tm + v + (kg << 3);
        int m1 = m0 + 16;
        out[((size_t)b * Lx + m0) * Cc + tn + l15]      = acc00[v] + bias0;
        out[((size_t)b * Lx + m0) * Cc + tn + 16 + l15] = acc01[v] + bias1;
        out[((size_t)b * Lx + m1) * Cc + tn + l15]      = acc10[v] + bias0;
        out[((size_t)b * Lx + m1) * Cc + tn + 16 + l15] = acc11[v] + bias1;
    }
}

// ================= InstanceNorm (+ReLU, +optional residual) =================
// src f32 [B,L,C]; per-(b,c) stats over L.
// pad_dst!=0: dst_h is halo-padded [B,18^3,C]; else flat [B,L,C].
__global__ void k_inorm_relu(const float* __restrict__ src,
                             _Float16* __restrict__ dst_h,
                             float* __restrict__ dst_f,
                             const float* __restrict__ resid,
                             int pad_dst) {
    int bc = blockIdx.x;                    // Bn*Cc blocks
    int b = bc / Cc, c = bc % Cc;
    int tid = threadIdx.x;                  // 256 threads
    const float* p = src + (size_t)b * Lx * Cc + c;
    float r[Lx / 256];
    float s = 0.f, s2 = 0.f;
#pragma unroll
    for (int i = 0; i < Lx / 256; ++i) {
        float v = p[(size_t)(tid + i * 256) * Cc];
        r[i] = v; s += v; s2 += v * v;
    }
    __shared__ float sA[256], sB[256];
    sA[tid] = s; sB[tid] = s2;
    __syncthreads();
    for (int st = 128; st > 0; st >>= 1) {
        if (tid < st) { sA[tid] += sA[tid + st]; sB[tid] += sB[tid + st]; }
        __syncthreads();
    }
    float mean = sA[0] * (1.0f / Lx);
    float var  = sB[0] * (1.0f / Lx) - mean * mean;
    float inv  = rsqrtf(var + 1e-5f);
#pragma unroll
    for (int i = 0; i < Lx / 256; ++i) {
        int l = tid + i * 256;
        float v = fmaxf((r[i] - mean) * inv, 0.f);
        if (resid) v += resid[((size_t)b * Cc + c) * Lx + l];
        if (dst_h) {
            if (pad_dst) {
                int dz = l >> 8, dy = (l >> 4) & 15, dx = l & 15;
                long pd = (long)(dz + 1) * PA + (dy + 1) * PR + (dx + 1);
                dst_h[((long)b * PV + pd) * Cc + c] = (_Float16)v;
            } else {
                dst_h[((size_t)b * Lx + l) * Cc + c] = (_Float16)v;
            }
        }
        if (dst_f) dst_f[((size_t)b * Lx + l) * Cc + c] = v;
    }
}

// ================= LayerNorm over C per token (wave per token) =================
__global__ void k_lnorm(const float* __restrict__ xg, const float* __restrict__ g,
                        const float* __restrict__ be, _Float16* __restrict__ tn) {
    int tok = blockIdx.x * 8 + (threadIdx.x >> 5);   // Mtok tokens, 8 waves/block
    int lane = threadIdx.x & 31;
    const float* p = xg + (size_t)tok * Cc;
    float r[Cc / 32];
    float s = 0.f, s2 = 0.f;
#pragma unroll
    for (int i = 0; i < Cc / 32; ++i) {
        float v = p[lane + i * 32];
        r[i] = v; s += v; s2 += v * v;
    }
    for (int off = 16; off > 0; off >>= 1) {
        s  += __shfl_xor(s,  off, 32);
        s2 += __shfl_xor(s2, off, 32);
    }
    float mean = s * (1.0f / Cc);
    float inv  = rsqrtf(s2 * (1.0f / Cc) - mean * mean + 1e-5f);
#pragma unroll
    for (int i = 0; i < Cc / 32; ++i) {
        int c = lane + i * 32;
        tn[(size_t)tok * Cc + c] = (_Float16)((r[i] - mean) * inv * g[c] + be[c]);
    }
}

// ================= causal depthwise conv1d + SiLU =================
__global__ void k_dwconv_silu(const float* __restrict__ xz, const float* __restrict__ w,
                              const float* __restrict__ bias,
                              float* __restrict__ uf, _Float16* __restrict__ uh) {
    long i = (long)blockIdx.x * blockDim.x + threadIdx.x;
    long n = (long)Mtok * DIN;
    if (i >= n) return;
    int d = (int)(i % DIN);
    long bl = i / DIN;
    int l = (int)(bl % Lx);
    long b = bl / Lx;
    float acc = bias[d];
#pragma unroll
    for (int j = 0; j < 4; ++j) {
        int ls = l - 3 + j;
        float xv = (ls >= 0) ? xz[((size_t)(b * Lx + ls)) * DIN2 + d] : 0.f;
        acc += xv * w[d * 4 + j];
    }
    float sv = acc * sigmoidf_(acc);
    uf[i] = sv;
    uh[i] = (_Float16)sv;
}

// ================= selective scan (serial over L, thread per (b,d)) =================
__global__ void k_scan(const float* __restrict__ dtv, const float* __restrict__ uf,
                       const float* __restrict__ dbl, const float* __restrict__ xz,
                       const float* __restrict__ An, const float* __restrict__ Dskip,
                       _Float16* __restrict__ yout) {
    int t = blockIdx.x * blockDim.x + threadIdx.x;
    if (t >= Bn * DIN) return;
    int b = t / DIN, d = t % DIN;
    float Af[DST], h[DST];
#pragma unroll
    for (int ns = 0; ns < DST; ++ns) { Af[ns] = An[d * DST + ns]; h[ns] = 0.f; }
    float ds = Dskip[d];
    for (int l = 0; l < Lx; ++l) {
        size_t base = (size_t)b * Lx + l;
        float dt = dtv[base * DIN + d];
        float uu = uf[base * DIN + d];
        float du = dt * uu;
        const float* row = dbl + base * DBLP;
        float y = 0.f;
#pragma unroll
        for (int ns = 0; ns < DST; ++ns) {
            float dA = __expf(dt * Af[ns]);
            h[ns] = dA * h[ns] + du * row[DTR + ns];
            y += h[ns] * row[DTR + DST + ns];
        }
        float z = xz[base * DIN2 + DIN + d];
        float yo = (y + uu * ds) * (z * sigmoidf_(z));
        yout[base * DIN + d] = (_Float16)yo;
    }
}

// ================= host orchestration =================
static inline long cdivl(long a, long b) { return (a + b - 1) / b; }

extern "C" void kernel_launch(void* const* d_in, const int* in_sizes, int n_in,
                              void* d_out, int out_size, void* d_ws, size_t ws_size,
                              hipStream_t stream) {
    (void)in_sizes; (void)n_in; (void)out_size; (void)ws_size;
    const float* x      = (const float*)d_in[0];
    const float* gw1    = (const float*)d_in[1];
    const float* gb1    = (const float*)d_in[2];
    const float* gw2    = (const float*)d_in[3];
    const float* gb2    = (const float*)d_in[4];
    const float* gw3    = (const float*)d_in[5];
    const float* gw4    = (const float*)d_in[7];
    const float* ln_g   = (const float*)d_in[9];
    const float* ln_b   = (const float*)d_in[10];
    const float* inpw   = (const float*)d_in[11];
    const float* c1w    = (const float*)d_in[12];
    const float* c1b    = (const float*)d_in[13];
    const float* xpw    = (const float*)d_in[14];
    const float* dtw    = (const float*)d_in[15];
    const float* dtb    = (const float*)d_in[16];
    const float* alog   = (const float*)d_in[17];
    const float* dskip  = (const float*)d_in[18];
    const float* outw   = (const float*)d_in[19];
    float* out = (float*)d_out;

    char* base = (char*)d_ws;
    auto alloc = [&](size_t bytes) -> void* {
        void* p = (void*)base;
        base += (bytes + 511) & ~((size_t)511);
        return p;
    };
    _Float16* xtf   = (_Float16*)alloc((size_t)Mtok * Cc * 2);       // flat [B,L,C]
    _Float16* xtp   = (_Float16*)alloc((size_t)Bn * PV * Cc * 2);    // padded conv1 in
    _Float16* x1p   = (_Float16*)alloc((size_t)Bn * PV * Cc * 2);    // padded conv2 in
    _Float16* wp1   = (_Float16*)alloc(27UL * Cc * Cc * 2);
    _Float16* wp2   = (_Float16*)alloc(27UL * Cc * Cc * 2);
    _Float16* w3h   = (_Float16*)alloc((size_t)Cc * Cc * 2);
    _Float16* w4h   = (_Float16*)alloc((size_t)Cc * Cc * 2);
    _Float16* inwh  = (_Float16*)alloc((size_t)DIN2 * Cc * 2);
    _Float16* xpwh  = (_Float16*)alloc((size_t)DBLP * DIN * 2);      // padded N=64
    _Float16* dtwh  = (_Float16*)alloc((size_t)DIN * DTRP * 2);      // padded K=32
    _Float16* outwh = (_Float16*)alloc((size_t)Cc * DIN * 2);
    float*    aneg  = (float*)   alloc((size_t)DIN * DST * 4);
    float*    convo = (float*)   alloc((size_t)Mtok * Cc * 4);
    _Float16* t1h   = (_Float16*)alloc((size_t)Mtok * Cc * 2);
    _Float16* x2h   = (_Float16*)alloc((size_t)Mtok * Cc * 2);
    _Float16* sh    = (_Float16*)alloc((size_t)Mtok * Cc * 2);
    float*    xg    = (float*)   alloc((size_t)Mtok * Cc * 4);
    _Float16* tnh   = (_Float16*)alloc((size_t)Mtok * Cc * 2);
    float*    xz    = (float*)   alloc((size_t)Mtok * DIN2 * 4);
    float*    uf    = (float*)   alloc((size_t)Mtok * DIN * 4);
    _Float16* uh    = (_Float16*)alloc((size_t)Mtok * DIN * 2);
    float*    dblf  = (float*)   alloc((size_t)Mtok * DBLP * 4);     // padded stride 64
    _Float16* dtinh = (_Float16*)alloc((size_t)Mtok * DTRP * 2);     // padded K=32
    float*    dtf   = (float*)   alloc((size_t)Mtok * DIN * 4);
    _Float16* youth = (_Float16*)alloc((size_t)Mtok * DIN * 2);

    const int TB = 256;
    // ---- zero halos, pack weights, cast input ----
    long padn = (long)Bn * PV * Cc;
    k_zero_h<<<cdivl(padn, TB), TB, 0, stream>>>(xtp, padn);
    k_zero_h<<<cdivl(padn, TB), TB, 0, stream>>>(x1p, padn);
    k_cast_x<<<cdivl((long)Bn * Cc * Lx, TB), TB, 0, stream>>>(x, xtf, xtp);
    k_pack_conv<<<cdivl(27L * Cc * Cc, TB), TB, 0, stream>>>(gw1, wp1);
    k_pack_conv<<<cdivl(27L * Cc * Cc, TB), TB, 0, stream>>>(gw2, wp2);
    k_cast_h<<<cdivl((long)Cc * Cc, TB), TB, 0, stream>>>(gw3, w3h, (long)Cc * Cc);
    k_cast_h<<<cdivl((long)Cc * Cc, TB), TB, 0, stream>>>(gw4, w4h, (long)Cc * Cc);
    k_cast_h<<<cdivl((long)DIN2 * Cc, TB), TB, 0, stream>>>(inpw, inwh, (long)DIN2 * Cc);
    k_pad_w<<<cdivl((long)DBLP * DIN, TB), TB, 0, stream>>>(xpw, xpwh, DBLW, DIN,
                                                            (long)DBLP * DIN, DIN);
    k_pad_w<<<cdivl((long)DIN * DTRP, TB), TB, 0, stream>>>(dtw, dtwh, DIN, DTR,
                                                            (long)DIN * DTRP, DTRP);
    k_cast_h<<<cdivl((long)Cc * DIN, TB), TB, 0, stream>>>(outw, outwh, (long)Cc * DIN);
    k_aneg<<<cdivl((long)DIN * DST, TB), TB, 0, stream>>>(alog, aneg, DIN * DST);

    // ---- GSC block ----
    long conv_waves = (long)Bn * (Lx / 32) * (Cc / 32);
    long conv_blocks = cdivl(conv_waves * 32, TB);
    // conv1 3x3x3 -> inorm/relu -> x1 (padded for conv2)
    k_conv3_wmma<<<conv_blocks, TB, 0, stream>>>(xtp, wp1, gb1, convo);
    k_inorm_relu<<<Bn * Cc, TB, 0, stream>>>(convo, x1p, nullptr, nullptr, 1);
    // conv2 3x3x3 -> inorm/relu -> t1 (flat)
    k_conv3_wmma<<<conv_blocks, TB, 0, stream>>>(x1p, wp2, gb2, convo);
    k_inorm_relu<<<Bn * Cc, TB, 0, stream>>>(convo, t1h, nullptr, nullptr, 0);
    // conv3 1x1x1 (GEMM) -> inorm/relu -> x2
    long g1_waves = (long)(Mtok / 32) * (Cc / 32);
    k_gemm_wmma<<<cdivl(g1_waves * 32, TB), TB, 0, stream>>>(xtf, Cc, w3h, Cc, convo, Cc,
                                                             Mtok, Cc, Cc, 0);
    k_inorm_relu<<<Bn * Cc, TB, 0, stream>>>(convo, x2h, nullptr, nullptr, 0);
    // s = t1 + x2 ; conv4 1x1x1 -> inorm/relu + residual x -> xg (f32)
    k_add_h<<<cdivl((long)Mtok * Cc, TB), TB, 0, stream>>>(t1h, x2h, sh, (long)Mtok * Cc);
    k_gemm_wmma<<<cdivl(g1_waves * 32, TB), TB, 0, stream>>>(sh, Cc, w4h, Cc, convo, Cc,
                                                             Mtok, Cc, Cc, 0);
    k_inorm_relu<<<Bn * Cc, TB, 0, stream>>>(convo, nullptr, xg, x, 0);

    // ---- Mamba layer ----
    k_lnorm<<<Mtok / 8, TB, 0, stream>>>(xg, ln_g, ln_b, tnh);
    // in_proj: [Mtok,384] x [1536,384]^T -> xz
    long g2_waves = (long)(Mtok / 32) * (DIN2 / 32);
    k_gemm_wmma<<<cdivl(g2_waves * 32, TB), TB, 0, stream>>>(tnh, Cc, inwh, Cc, xz, DIN2,
                                                             Mtok, DIN2, Cc, 0);
    // depthwise causal conv1d + SiLU -> u
    k_dwconv_silu<<<cdivl((long)Mtok * DIN, TB), TB, 0, stream>>>(xz, c1w, c1b, uf, uh);
    // x_proj: [Mtok,768] x [64,768]^T -> dbl (stride 64)
    long g3_waves = (long)(Mtok / 32) * (DBLP / 32);
    k_gemm_wmma<<<cdivl(g3_waves * 32, TB), TB, 0, stream>>>(uh, DIN, xpwh, DIN, dblf, DBLP,
                                                             Mtok, DBLP, DIN, 0);
    // dt_proj: [Mtok,32pad] x [768,32pad]^T -> dt ; then softplus(+bias)
    k_slice_dt<<<cdivl((long)Mtok * DTRP, TB), TB, 0, stream>>>(dblf, dtinh);
    long g4_waves = (long)(Mtok / 32) * (DIN / 32);
    k_gemm_wmma<<<cdivl(g4_waves * 32, TB), TB, 0, stream>>>(dtinh, DTRP, dtwh, DTRP, dtf, DIN,
                                                             Mtok, DIN, DTRP, 0);
    k_softplus_bias<<<cdivl((long)Mtok * DIN, TB), TB, 0, stream>>>(dtf, dtb);
    // selective scan + skip + gate -> yout (f16)
    k_scan<<<cdivl((long)Bn * DIN, TB), TB, 0, stream>>>(dtf, uf, dblf, xz, aneg, dskip, youth);
    // out_proj: [Mtok,768] x [384,768]^T -> out in [B,C,D,H,W] layout
    long g5_waves = (long)(Mtok / 32) * (Cc / 32);
    k_gemm_wmma<<<cdivl(g5_waves * 32, TB), TB, 0, stream>>>(youth, DIN, outwh, DIN, out, Cc,
                                                             Mtok, Cc, DIN, 1);
}
